// MultiHeadedAttention1_9294309229326
// MI455X (gfx1250) — compile-verified, matrix operand order checked by reading, not run on hardware
//
#include <hip/hip_runtime.h>

#define B_    4
#define C_    1024
#define D_    1024
#define H_    16
#define DH_   64
#define KTOP_ 716

typedef __attribute__((ext_vector_type(16))) __bf16 bf16x16;
typedef __attribute__((ext_vector_type(8)))  float  f32x8;
typedef __attribute__((ext_vector_type(4)))  unsigned int u32x4;
typedef __attribute__((ext_vector_type(8)))  unsigned int u32x8;

union FragBF { bf16x16 v; uint4 u[2]; unsigned short s[16]; };
union U4BF   { uint4 q; unsigned short s[8]; };

__device__ inline unsigned short f2bf(float f) {
  unsigned u = __float_as_uint(f);
  unsigned r = ((u >> 16) & 1u) + 0x7FFFu;   // round-to-nearest-even
  return (unsigned short)((u + r) >> 16);
}
// order-preserving float -> uint map (larger float => larger uint)
__device__ inline unsigned f2ord(float f) {
  unsigned u = __float_as_uint(f);
  return (u & 0x80000000u) ? ~u : (u | 0x80000000u);
}
__device__ inline int   wsum_i(int v)   { for (int o = 16; o > 0; o >>= 1) v += __shfl_xor(v, o, 32); return v; }
__device__ inline float wsum_f(float v) { for (int o = 16; o > 0; o >>= 1) v += __shfl_xor(v, o, 32); return v; }
__device__ inline float wmax_f(float v) { for (int o = 16; o > 0; o >>= 1) v = fmaxf(v, __shfl_xor(v, o, 32)); return v; }

// ---------------------------------------------------------- TDM tile load ---
// Issue a Tensor Data Mover 2-D tile load (global -> LDS), descriptor per
// CDNA5 ISA. One call = one DMA of tile_w x tile_h
// elements (2 bytes each) with optional LDS row padding. Wave-level op;
// EXEC ignored. Completion tracked with TENSORcnt.
__device__ inline void tdm_load_2d_lds(const void* lds_dst, const void* gsrc,
                                       unsigned tensor_w, unsigned tensor_h,
                                       unsigned tile_w, unsigned tile_h,
                                       unsigned row_stride_elems,
                                       unsigned pad_int_code, unsigned pad_amt_code)
{
  unsigned long long ga = (unsigned long long)(size_t)gsrc;
  unsigned lo = (unsigned)(size_t)lds_dst;           // low 32 bits = LDS byte offset
  u32x4 g0;
  g0[0] = 1u;                                        // count=1, user descriptor
  g0[1] = lo;                                        // lds_addr
  g0[2] = (unsigned)ga;                              // global_addr[31:0]
  g0[3] = ((unsigned)(ga >> 32) & 0x01ffffffu)       // global_addr[56:32]
          | (2u << 30);                              // type = 2 ("image")
  u32x8 g1;
  g1[0] = (1u << 16)                                 // data_size = 2 bytes
        | (1u << 20)                                 // pad_enable
        | (pad_int_code << 22) | (pad_amt_code << 25);
  g1[1] = (tensor_w & 0xffffu) << 16;                              // tensor_dim0 lo16 (bits 63:48)
  g1[2] = ((tensor_w >> 16) & 0xffffu) | ((tensor_h & 0xffffu) << 16); // dim0 hi | dim1 lo
  g1[3] = ((tensor_h >> 16) & 0xffffu) | ((tile_w & 0xffffu) << 16);   // dim1 hi | tile_dim0
  g1[4] = (tile_h & 0xffffu);                        // tile_dim1 (tile_dim2 = 0)
  g1[5] = row_stride_elems;                          // tensor_dim0_stride[31:0]
  g1[6] = 0u;                                        // stride hi16 | dim1_stride lo16
  g1[7] = 0u;
  asm volatile("tensor_load_to_lds %0, %1" :: "s"(g0), "s"(g1) : "memory");
}

// ---------------------------------------------------------------- convert ---
__global__ void cvt_f32_to_bf16(const float* __restrict__ in,
                                unsigned short* __restrict__ out, int n) {
  int i = blockIdx.x * blockDim.x + threadIdx.x;
  if (i < n) out[i] = f2bf(in[i]);
}

// ------------------------------------------------------------------- GEMM ---
// Y[M=4096, N=1024] = X[4096,1024] @ W^T (+bias)*scale, bf16 in / f32 acc.
// Tiles staged by TDM with double buffering; LDS rows padded to 48 elems
// (96B, multiple of 16B) by TDM pad feature: 16 DWORDs data + 8 DWORDs pad.
// mode 0: split-heads bf16 out [B,H,C,DH] ; mode 2: f32 out [M,N].
__global__ __launch_bounds__(256) void gemm_bf16_wmma(
    const unsigned short* __restrict__ A,
    const unsigned short* __restrict__ W,
    const float* __restrict__ bias,
    unsigned short* __restrict__ outBF,
    float* __restrict__ outF,
    float scale, int mode, int nrowsA)
{
  __shared__ __align__(16) unsigned short Ash[2][128 * 48];
  __shared__ __align__(16) unsigned short Bsh[2][128 * 48];

  const int tid  = threadIdx.x;
  const int lane = tid & 31, wid = tid >> 5;
  const int wm = wid >> 2, wn = wid & 3;               // 2x4 wave grid
  const int bn = blockIdx.x * 128, bm = blockIdx.y * 128;
  const int lrow = lane & 15, kb = (lane >> 4) * 8;

  const f32x8 zero8 = {0.f,0.f,0.f,0.f,0.f,0.f,0.f,0.f};
  f32x8 acc[4][2];
  for (int i = 0; i < 4; ++i) for (int j = 0; j < 2; ++j) acc[i][j] = zero8;

  const int NIT = D_ / 32;
  if (wid == 0) {   // prime the pipeline: tiles for k0 = 0 into buffer 0
    tdm_load_2d_lds(&Ash[0][0], &A[(size_t)bm * D_], D_, (unsigned)nrowsA, 32, 128, D_, 3, 7);
    tdm_load_2d_lds(&Bsh[0][0], &W[(size_t)bn * D_], D_, D_,               32, 128, D_, 3, 7);
  }

  for (int it = 0; it < NIT; ++it) {
    __builtin_amdgcn_s_wait_tensorcnt(0);   // issuer waits its DMAs; others pass
    __syncthreads();                        // tile `it` now visible to all waves
    const int buf = it & 1;
    if (wid == 0 && it + 1 < NIT) {         // stream next tile while computing
      int k0n = (it + 1) * 32;
      tdm_load_2d_lds(&Ash[buf ^ 1][0], &A[(size_t)bm * D_ + k0n], D_, (unsigned)nrowsA, 32, 128, D_, 3, 7);
      tdm_load_2d_lds(&Bsh[buf ^ 1][0], &W[(size_t)bn * D_ + k0n], D_, D_,               32, 128, D_, 3, 7);
    }

    FragBF af[4], bf[2];
    for (int i = 0; i < 4; ++i) {
      int r = wm * 64 + i * 16 + lrow;
      af[i].u[0] = *(const uint4*)&Ash[buf][r * 48 + kb];
      af[i].u[1] = *(const uint4*)&Ash[buf][r * 48 + kb + 16];
    }
    for (int j = 0; j < 2; ++j) {
      int n = wn * 32 + j * 16 + lrow;
      bf[j].u[0] = *(const uint4*)&Bsh[buf][n * 48 + kb];
      bf[j].u[1] = *(const uint4*)&Bsh[buf][n * 48 + kb + 16];
    }
    for (int i = 0; i < 4; ++i)
      for (int j = 0; j < 2; ++j)
        acc[i][j] = __builtin_amdgcn_wmma_f32_16x16x32_bf16(
            false, af[i].v, false, bf[j].v, (short)0, acc[i][j], false, false);
  }

  // epilogue: bias + scale, layout per mode
  for (int i = 0; i < 4; ++i)
    for (int j = 0; j < 2; ++j) {
      int n = bn + wn * 32 + j * 16 + lrow;
      float bv = bias[n];
      for (int e = 0; e < 8; ++e) {
        int r = bm + wm * 64 + i * 16 + kb + e;
        float y = (acc[i][j][e] + bv) * scale;
        if (mode == 0) {
          int b = r >> 10, c = r & 1023, h = n >> 6, dh = n & 63;
          outBF[(((size_t)(b * H_ + h)) * C_ + c) * DH_ + dh] = f2bf(y);
        } else {
          outF[(size_t)r * D_ + n] = y;
        }
      }
    }
}

// -------------------------------------------------------------- attention ---
// One block: 32 query rows for one (b,h). Scores 32x1024 f32 live in LDS
// (132 KB of the 320 KB WGP LDS). Q tile staged by TDM.
__global__ __launch_bounds__(256) void attn_topk_wmma(
    const unsigned short* __restrict__ Qp,   // [B*H, C, DH] bf16
    const unsigned short* __restrict__ Kp,
    const unsigned short* __restrict__ Vp,
    unsigned short* __restrict__ Ctx)        // [B, C, H, DH] bf16
{
  extern __shared__ char smem[];
  unsigned short* Qs = (unsigned short*)smem;                    // 32 x 72
  unsigned short* Vt = (unsigned short*)(smem + 32 * 72 * 2);    // 64 x 48 (V^T tile)
  float*          S  = (float*)(smem + 32 * 72 * 2 + 64 * 48 * 2); // 32 x 1032

  const int tid = threadIdx.x, lane = tid & 31, wid = tid >> 5;
  const int bh = blockIdx.x;
  const int q0 = blockIdx.y * 32;
  const int lrow = lane & 15, kb8 = (lane >> 4) * 8;
  const size_t base = (size_t)bh * C_ * DH_;
  const f32x8 zero8 = {0.f,0.f,0.f,0.f,0.f,0.f,0.f,0.f};

  // stage Q tile (32 x 64 bf16) via TDM; pad 32 DW data + 4 DW -> stride 72
  if (wid == 0)
    tdm_load_2d_lds(Qs, &Qp[base + (size_t)q0 * DH_], DH_, C_, DH_, 32, DH_, 4, 3);
  __builtin_amdgcn_s_wait_tensorcnt(0);
  __syncthreads();

  // --- phase 1: S = Q @ K^T via WMMA (each wave owns 8 key-tiles) ---
  for (int t = 0; t < 8; ++t) {
    int ntile = wid * 8 + t;
    int n = ntile * 16 + lrow;
    f32x8 a0 = zero8, a1 = zero8;
    for (int ks = 0; ks < DH_; ks += 32) {
      int kk = ks + kb8;
      FragBF kf, qf0, qf1;
      const uint4* g = (const uint4*)&Kp[base + (size_t)n * DH_ + kk];
      kf.u[0] = g[0]; kf.u[1] = g[2];
      qf0.u[0] = *(const uint4*)&Qs[lrow * 72 + kk];
      qf0.u[1] = *(const uint4*)&Qs[lrow * 72 + kk + 16];
      qf1.u[0] = *(const uint4*)&Qs[(16 + lrow) * 72 + kk];
      qf1.u[1] = *(const uint4*)&Qs[(16 + lrow) * 72 + kk + 16];
      a0 = __builtin_amdgcn_wmma_f32_16x16x32_bf16(false, qf0.v, false, kf.v, (short)0, a0, false, false);
      a1 = __builtin_amdgcn_wmma_f32_16x16x32_bf16(false, qf1.v, false, kf.v, (short)0, a1, false, false);
    }
    if (t < 7) __builtin_prefetch(&Kp[base + (size_t)(n + 16) * DH_], 0, 1);
    int col = ntile * 16 + lrow;
    for (int e = 0; e < 8; ++e) {
      S[(kb8 + e) * 1032 + col]      = a0[e];
      S[(16 + kb8 + e) * 1032 + col] = a1[e];
    }
  }
  __syncthreads();

  // --- phase 2: exact top-K threshold + masked softmax (4 rows per wave) ---
  for (int rr = 0; rr < 4; ++rr) {
    int row = wid * 4 + rr;
    float* Sr = &S[row * 1032];

    unsigned tsel = 0u;                       // max t with count(ord >= t) >= K
    for (int bit = 31; bit >= 0; --bit) {
      unsigned cand = tsel | (1u << bit);
      int cnt = 0;
      for (int i = 0; i < 32; ++i)
        cnt += (f2ord(Sr[lane + 32 * i]) >= cand) ? 1 : 0;
      cnt = wsum_i(cnt);
      if (cnt >= KTOP_) tsel = cand;
    }

    float m = -3.4e38f;
    for (int i = 0; i < 32; ++i) m = fmaxf(m, Sr[lane + 32 * i]);
    m = wmax_f(m);
    float sum = 0.f;
    for (int i = 0; i < 32; ++i) {
      float v = Sr[lane + 32 * i];
      if (f2ord(v) >= tsel) sum += __expf(v - m);
    }
    sum = wsum_f(sum);
    float inv = 1.0f / sum;
    for (int i = 0; i < 32; ++i) {
      int idx = lane + 32 * i;
      float v = Sr[idx];
      Sr[idx] = (f2ord(v) >= tsel) ? __expf(v - m) * inv : 0.0f;
    }
  }
  __syncthreads();

  // --- phase 3: ctx = attn @ V via WMMA (wave -> one 16x16 output tile) ---
  const int mt = wid >> 2, nt = wid & 3;
  f32x8 acc = zero8;
  for (int kt = 0; kt < 32; ++kt) {
    { // stage V^T tile: 32 keys x 64 dh -> Vt[dh][key]
      int key = tid >> 3, cc = (tid & 7) << 3;
      U4BF vrow;
      vrow.q = *(const uint4*)&Vp[base + (size_t)(kt * 32 + key) * DH_ + cc];
      if (kt + 1 < 32)
        __builtin_prefetch(&Vp[base + (size_t)((kt + 1) * 32 + key) * DH_ + cc], 0, 1);
      for (int e = 0; e < 8; ++e) Vt[(cc + e) * 48 + key] = vrow.s[e];
    }
    __syncthreads();
    FragBF af, bf;
    {
      int m = mt * 16 + lrow;
      const float* sr = &S[m * 1032 + kt * 32 + kb8];
      for (int e = 0; e < 8; ++e) { af.s[e] = f2bf(sr[e]); af.s[8 + e] = f2bf(sr[16 + e]); }
      int n = nt * 16 + lrow;
      bf.u[0] = *(const uint4*)&Vt[n * 48 + kb8];
      bf.u[1] = *(const uint4*)&Vt[n * 48 + kb8 + 16];
    }
    acc = __builtin_amdgcn_wmma_f32_16x16x32_bf16(false, af.v, false, bf.v, (short)0, acc, false, false);
    __syncthreads();
  }

  { // store ctx as [B, C, H, DH] bf16 (row-major [B*C, D] for final GEMM)
    int b = bh >> 4, h = bh & 15;
    int dh = nt * 16 + lrow;
    for (int e = 0; e < 8; ++e) {
      int c = q0 + mt * 16 + kb8 + e;
      Ctx[(((size_t)b * C_ + c) * H_ + h) * DH_ + dh] = f2bf(acc[e]);
    }
  }
}

// ------------------------------------------------------------------ launch ---
extern "C" void kernel_launch(void* const* d_in, const int* in_sizes, int n_in,
                              void* d_out, int out_size, void* d_ws, size_t ws_size,
                              hipStream_t stream) {
  const float* key   = (const float*)d_in[0];
  const float* value = (const float*)d_in[1];
  const float* query = (const float*)d_in[2];
  const float* Wq = (const float*)d_in[3];
  const float* bq = (const float*)d_in[4];
  const float* Wk = (const float*)d_in[5];
  const float* bk = (const float*)d_in[6];
  const float* Wv = (const float*)d_in[7];
  const float* bv = (const float*)d_in[8];
  const float* Wo = (const float*)d_in[9];
  const float* bo = (const float*)d_in[10];
  float* out = (float*)d_out;

  const size_t NTOK = (size_t)B_ * C_;          // 4096 rows
  unsigned short* p = (unsigned short*)d_ws;
  unsigned short* qbf = p;           p += NTOK * D_;
  unsigned short* kbf = p;           p += NTOK * D_;
  unsigned short* vbf = p;           p += NTOK * D_;
  unsigned short* wqb = p;           p += (size_t)D_ * D_;
  unsigned short* wkb = p;           p += (size_t)D_ * D_;
  unsigned short* wvb = p;           p += (size_t)D_ * D_;
  unsigned short* wob = p;           p += (size_t)D_ * D_;
  unsigned short* Qp  = p;           p += NTOK * D_;
  unsigned short* Kp  = p;           p += NTOK * D_;
  unsigned short* Vp  = p;           p += NTOK * D_;
  unsigned short* Ctx = p;           p += NTOK * D_;

  const int CV = 256;
  const int nTok = (int)(NTOK * D_);            // 4 Mi elements
  const int nW = D_ * D_;                       // 1 Mi elements
  cvt_f32_to_bf16<<<(nTok + CV - 1) / CV, CV, 0, stream>>>(query, qbf, nTok);
  cvt_f32_to_bf16<<<(nTok + CV - 1) / CV, CV, 0, stream>>>(key,   kbf, nTok);
  cvt_f32_to_bf16<<<(nTok + CV - 1) / CV, CV, 0, stream>>>(value, vbf, nTok);
  cvt_f32_to_bf16<<<(nW + CV - 1) / CV, CV, 0, stream>>>(Wq, wqb, nW);
  cvt_f32_to_bf16<<<(nW + CV - 1) / CV, CV, 0, stream>>>(Wk, wkb, nW);
  cvt_f32_to_bf16<<<(nW + CV - 1) / CV, CV, 0, stream>>>(Wv, wvb, nW);
  cvt_f32_to_bf16<<<(nW + CV - 1) / CV, CV, 0, stream>>>(Wo, wob, nW);

  dim3 ggrid(D_ / 128, (int)(NTOK / 128));      // (8, 32)
  const float qscale = 0.125f;                  // 1/sqrt(DH)
  gemm_bf16_wmma<<<ggrid, 256, 0, stream>>>(qbf, wqb, bq, Qp, nullptr, qscale, 0, (int)NTOK);
  gemm_bf16_wmma<<<ggrid, 256, 0, stream>>>(kbf, wkb, bk, Kp, nullptr, 1.0f, 0, (int)NTOK);
  gemm_bf16_wmma<<<ggrid, 256, 0, stream>>>(vbf, wvb, bv, Vp, nullptr, 1.0f, 0, (int)NTOK);

  size_t smem = (size_t)32 * 72 * 2 + (size_t)64 * 48 * 2 + (size_t)32 * 1032 * 4;
  attn_topk_wmma<<<dim3(B_ * H_, C_ / 32), 256, smem, stream>>>(Qp, Kp, Vp, Ctx);

  gemm_bf16_wmma<<<ggrid, 256, 0, stream>>>(Ctx, wob, bo, nullptr, out, 1.0f, 2, (int)NTOK);
}
